// GCN_Convlution_55877524521314
// MI455X (gfx1250) — matrix operand order repaired and checked by reading, not compiled, not run on hardware
//
#include <hip/hip_runtime.h>

// Problem constants (match reference)
#define BQ   128      // batch
#define LQ   216      // seq len
#define LP   224      // padded seq len (multiple of 16 and 32)
#define EQ   256      // embedding dim
#define UQ   256      // units
#define LEVQ 8        // levels
#define CHQ  32       // chunk per level
#define GD   1040     // 4*U + 2*LEV

typedef __attribute__((ext_vector_type(16))) __bf16 v16bf;
typedef __attribute__((ext_vector_type(8)))  float  v8f;

__device__ __forceinline__ unsigned short f2bf(float f) {
  unsigned int u = __float_as_uint(f);
  return (unsigned short)((u + 0x7FFFu + ((u >> 16) & 1u)) >> 16);
}
__device__ __forceinline__ float bf2f(unsigned short s) {
  return __uint_as_float(((unsigned int)s) << 16);
}
__device__ __forceinline__ float sigm(float x) { return 1.0f / (1.0f + __expf(-x)); }

union FragU { v16bf v; uint4 q[2]; };

// A fragment (16x32 bf16, row-major A). Lane<16: row=m0+lane, K base k0;
// lane>=16: same rows, K base k0+8. Each half-lane holds K {kb..kb+7} U {kb+16..kb+23}.
__device__ __forceinline__ v16bf load_frag_a(const unsigned short* A, int lda,
                                             int m0, int k0, int lane) {
  int r  = m0 + (lane & 15);
  int kb = k0 + ((lane >> 4) << 3);
  const unsigned short* p = A + (size_t)r * lda + kb;
  FragU u;
  u.q[0] = *(const uint4*)(p);
  u.q[1] = *(const uint4*)(p + 16);
  return u.v;
}

// B fragment (32x16 bf16) from TRANSPOSED weight Bt[N][K]. Lane<16: n=n0+lane,
// K base k0; lane>=16: K base k0+8. Each half-lane holds K {kb..kb+7} U {kb+16..kb+23}.
__device__ __forceinline__ v16bf load_frag_b(const unsigned short* Bt, int ldb,
                                             int n0, int k0, int lane) {
  int n  = n0 + (lane & 15);
  int kb = k0 + ((lane >> 4) << 3);
  const unsigned short* p = Bt + (size_t)n * ldb + kb;
  FragU u;
  u.q[0] = *(const uint4*)(p);
  u.q[1] = *(const uint4*)(p + 16);
  return u.v;
}

// ---------------- prep kernels ----------------

// out[n][k] = bf16(in[k][n]); in is [Kd][Nd] row-major f32.
__global__ void transpose_bf16_kernel(const float* __restrict__ in,
                                      unsigned short* __restrict__ out,
                                      int Kd, int Nd) {
  int idx = blockIdx.x * 256 + threadIdx.x;
  if (idx >= Nd * Kd) return;
  int n = idx / Kd;
  int k = idx - n * Kd;
  out[idx] = f2bf(in[(size_t)k * Nd + n]);
}

__global__ void embed_kernel(const int* __restrict__ ids,
                             const float* __restrict__ emb,
                             unsigned short* __restrict__ x0) {
  size_t idx = (size_t)blockIdx.x * 256 + threadIdx.x;
  if (idx >= (size_t)BQ * LP * EQ) return;
  int e = (int)(idx % EQ);
  int l = (int)((idx / EQ) % LP);
  int b = (int)(idx / ((size_t)EQ * LP));
  x0[idx] = (l < LQ) ? f2bf(emb[(size_t)ids[b * LQ + l] * EQ + e]) : (unsigned short)0;
}

__global__ void pad_graphs_kernel(const float* __restrict__ g,
                                  unsigned short* __restrict__ out) {
  size_t idx = (size_t)blockIdx.x * 256 + threadIdx.x;
  if (idx >= (size_t)BQ * LP * LP) return;
  int j = (int)(idx % LP);
  int i = (int)((idx / LP) % LP);
  int b = (int)(idx / ((size_t)LP * LP));
  out[idx] = (i < LQ && j < LQ) ? f2bf(g[((size_t)b * LQ + i) * LQ + j]) : (unsigned short)0;
}

__global__ void zero_u16_kernel(unsigned short* __restrict__ p, size_t n) {
  size_t idx = (size_t)blockIdx.x * 256 + threadIdx.x;
  if (idx < n) p[idx] = 0;
}

// ---------------- generic (batched) WMMA GEMM: C = A(bf16) * Bt^T(bf16) + bias ----------------
// A:  [M x K] row-major bf16 (lda), Bt: [N x K] row-major bf16 (ldb, pre-transposed weight)
// Output bf16 [M x N] (ldc). Grid: (M/16, ceil(Ntiles/4), batches). Block: 128 (4 waves).
__global__ __launch_bounds__(128)
void wmma_gemm_kernel(const unsigned short* __restrict__ A, int lda, size_t strideA,
                      const unsigned short* __restrict__ Bt, int ldb, size_t strideB,
                      const float* __restrict__ bias,
                      unsigned short* __restrict__ C, int ldc, size_t strideC,
                      int Ntiles, int K) {
  int lane = threadIdx.x & 31;
  int wave = threadIdx.x >> 5;
  int nt = blockIdx.y * 4 + wave;
  if (nt >= Ntiles) return;          // wave-uniform: EXEC stays all-ones for WMMA
  int m0 = blockIdx.x * 16;
  int n0 = nt * 16;
  A  += (size_t)blockIdx.z * strideA;
  Bt += (size_t)blockIdx.z * strideB;
  int col = n0 + (lane & 15);
  float binit = bias ? bias[col] : 0.0f;
  v8f acc;
#pragma unroll
  for (int i = 0; i < 8; ++i) acc[i] = binit;
  for (int k0 = 0; k0 < K; k0 += 32) {
    v16bf a = load_frag_a(A, lda, m0, k0, lane);
    v16bf b = load_frag_b(Bt, ldb, n0, k0, lane);
    acc = __builtin_amdgcn_wmma_f32_16x16x32_bf16(false, a, false, b,
                                                  (short)0, acc, false, false);
  }
  unsigned short* out = C + (size_t)blockIdx.z * strideC;
  int row = m0 + ((lane >> 4) << 3);
#pragma unroll
  for (int i = 0; i < 8; ++i) out[(size_t)(row + i) * ldc + col] = f2bf(acc[i]);
}

// ---------------- ON-LSTM recurrent scan (persistent per 16 batch rows) ----------------
// xk:  [B][LP][GD] bf16 precomputed input projection (+bias)
// rkt: [GD][U] bf16 transposed recurrent weight
// seq/seqT: optional bf16 outputs [B][LP][U] and [B][U][LP]; hfinal: optional f32 [B][512]
__global__ __launch_bounds__(256)
void onlstm_scan_kernel(const unsigned short* __restrict__ xk,
                        const unsigned short* __restrict__ rkt,
                        unsigned short* __restrict__ seq,
                        unsigned short* __restrict__ seqT,
                        float* __restrict__ hfinal) {
  __shared__ __attribute__((aligned(16))) unsigned short h_s[16 * UQ]; // 8 KB
  __shared__ __attribute__((aligned(16))) float          c_s[16 * UQ]; // 16 KB
  __shared__ __attribute__((aligned(16))) unsigned short z_s[16 * GD]; // 32.5 KB
  int tid  = threadIdx.x;
  int lane = tid & 31;
  int wave = tid >> 5;
  int b0 = blockIdx.x * 16;
  for (int i = tid; i < 16 * UQ; i += 256) { h_s[i] = 0; c_s[i] = 0.0f; }
  __syncthreads();
  const size_t rowstride = (size_t)LP * GD;   // batch-row stride in xk
  for (int t = 0; t < LQ; ++t) {
    const unsigned short* xkt = xk + ((size_t)b0 * LP + t) * GD;
    int rb = (lane >> 4) << 3;
    // Phase 1: z = xk[t] + h @ rkernel.
    // h is fixed within a timestep: load all 8 K-fragments of h ONCE (64 VGPRs)
    // and reuse them across every N tile this wave owns (8x fewer LDS reads).
    v16bf afrag[8];
#pragma unroll
    for (int kk = 0; kk < 8; ++kk)
      afrag[kk] = load_frag_a(h_s, UQ, 0, kk * 32, lane);   // ds_load_b128 x2
    for (int nt = wave; nt < GD / 16; nt += 8) {
      int n0 = nt * 16;
      int col = n0 + (lane & 15);
      if (nt + 8 < GD / 16)
        __builtin_prefetch(rkt + (size_t)(col + 128) * UQ, 0, 0); // global_prefetch_b8
      v8f acc;
#pragma unroll
      for (int i = 0; i < 8; ++i)
        acc[i] = bf2f(xkt[(size_t)(rb + i) * rowstride + col]);
#pragma unroll
      for (int kk = 0; kk < 8; ++kk) {
        v16bf b = load_frag_b(rkt, UQ, n0, kk * 32, lane);  // L2-resident weight
        acc = __builtin_amdgcn_wmma_f32_16x16x32_bf16(false, afrag[kk], false, b,
                                                      (short)0, acc, false, false);
      }
#pragma unroll
      for (int i = 0; i < 8; ++i) z_s[(rb + i) * GD + col] = f2bf(acc[i]);
    }
    __syncthreads();
    // Phase 2: gates. Thread layout: row r = tid>>4 (16 rows), 16 units per thread.
    int r = tid >> 4;
    float fm[LEVQ], im[LEVQ];
    {
      float zv[LEVQ]; float mx = -3.0e38f;
#pragma unroll
      for (int j = 0; j < LEVQ; ++j) { zv[j] = bf2f(z_s[r * GD + j]); mx = fmaxf(mx, zv[j]); }
      float s = 0.0f;
#pragma unroll
      for (int j = 0; j < LEVQ; ++j) { zv[j] = __expf(zv[j] - mx); s += zv[j]; }
      float inv = 1.0f / s, cum = 0.0f;
#pragma unroll
      for (int j = 0; j < LEVQ; ++j) { cum += zv[j] * inv; fm[j] = cum; } // prefix-sum softmax
    }
    {
      float zv[LEVQ]; float mx = -3.0e38f;
#pragma unroll
      for (int j = 0; j < LEVQ; ++j) { zv[j] = bf2f(z_s[r * GD + LEVQ + j]); mx = fmaxf(mx, zv[j]); }
      float s = 0.0f;
#pragma unroll
      for (int j = 0; j < LEVQ; ++j) { zv[j] = __expf(zv[j] - mx); s += zv[j]; }
      float inv = 1.0f / s, cum = 0.0f;
#pragma unroll
      for (int j = LEVQ - 1; j >= 0; --j) { cum += zv[j] * inv; im[j] = cum; } // suffix-sum softmax
    }
    int ub = (tid & 15) * 16;
#pragma unroll
    for (int j = 0; j < 16; ++j) {
      int u = ub + j;
      int lev = u >> 5;                           // CH = 32
      float fg = sigm(bf2f(z_s[r * GD + 16 + u]));
      float ig = sigm(bf2f(z_s[r * GD + 16 + 256 + u]));
      float og = sigm(bf2f(z_s[r * GD + 16 + 512 + u]));
      float ci = tanhf(bf2f(z_s[r * GD + 16 + 768 + u]));
      float c  = c_s[r * UQ + u];
      float ov = fm[lev] * im[lev];
      float cn = ov * (fg * c + ig * ci) + (fm[lev] - ov) * c + (im[lev] - ov) * ci;
      float hn = og * tanhf(cn);
      c_s[r * UQ + u] = cn;
      unsigned short hb = f2bf(hn);
      h_s[r * UQ + u] = hb;
      if (seq)  seq[((size_t)(b0 + r) * LP + t) * UQ + u] = hb;
      if (seqT) seqT[((size_t)(b0 + r) * UQ + u) * LP + t] = hb;
      if (hfinal && t == LQ - 1) hfinal[(size_t)(b0 + r) * (2 * UQ) + u] = hn;
    }
    __syncthreads();
  }
}

// ---------------- host orchestration ----------------

extern "C" void kernel_launch(void* const* d_in, const int* in_sizes, int n_in,
                              void* d_out, int out_size, void* d_ws, size_t ws_size,
                              hipStream_t stream) {
  (void)in_sizes; (void)n_in; (void)out_size; (void)ws_size;
  const int*   ids       = (const int*)d_in[0];
  const float* graphs    = (const float*)d_in[1];
  const float* emb       = (const float*)d_in[2];
  const float* kernels   = (const float*)d_in[3];
  const float* rkernels  = (const float*)d_in[4];
  const float* biases    = (const float*)d_in[5];
  const float* kernel_f  = (const float*)d_in[6];
  const float* rkernel_f = (const float*)d_in[7];
  const float* bias_f    = (const float*)d_in[8];
  const float* gtran     = (const float*)d_in[9];
  float* out = (float*)d_out;

  char* ws = (char*)d_ws;
  size_t off = 0;
  auto alloc = [&](size_t bytes) -> void* {
    void* p = ws + off;
    off = (off + bytes + 255) & ~(size_t)255;
    return p;
  };
  unsigned short* kt0  = (unsigned short*)alloc((size_t)GD * EQ * 2);
  unsigned short* kt1  = (unsigned short*)alloc((size_t)GD * EQ * 2);
  unsigned short* rkt0 = (unsigned short*)alloc((size_t)GD * UQ * 2);
  unsigned short* rkt1 = (unsigned short*)alloc((size_t)GD * UQ * 2);
  unsigned short* ktf  = (unsigned short*)alloc((size_t)GD * UQ * 2);
  unsigned short* rktf = (unsigned short*)alloc((size_t)GD * UQ * 2);
  unsigned short* gtt  = (unsigned short*)alloc((size_t)UQ * UQ * 2);
  unsigned short* x0   = (unsigned short*)alloc((size_t)BQ * LP * EQ * 2);
  unsigned short* x1   = (unsigned short*)alloc((size_t)BQ * LP * UQ * 2);
  unsigned short* x2   = (unsigned short*)alloc((size_t)BQ * LP * UQ * 2);
  unsigned short* x2T  = (unsigned short*)alloc((size_t)BQ * UQ * LP * 2);
  unsigned short* gpad = (unsigned short*)alloc((size_t)BQ * LP * LP * 2);
  unsigned short* g1   = (unsigned short*)alloc((size_t)BQ * LP * UQ * 2);
  unsigned short* gbf  = (unsigned short*)alloc((size_t)BQ * LP * UQ * 2);
  unsigned short* xkb  = (unsigned short*)alloc((size_t)BQ * LP * GD * 2);

  const float* bias0 = biases;
  const float* bias1 = biases + GD;

  // 1) weight transpose + bf16 conversion
  {
    int n = GD * EQ, g = (n + 255) / 256;
    transpose_bf16_kernel<<<g, 256, 0, stream>>>(kernels, kt0, EQ, GD);
    transpose_bf16_kernel<<<g, 256, 0, stream>>>(kernels + (size_t)EQ * GD, kt1, EQ, GD);
    transpose_bf16_kernel<<<g, 256, 0, stream>>>(rkernels, rkt0, UQ, GD);
    transpose_bf16_kernel<<<g, 256, 0, stream>>>(rkernels + (size_t)UQ * GD, rkt1, UQ, GD);
    transpose_bf16_kernel<<<g, 256, 0, stream>>>(kernel_f, ktf, UQ, GD);
    transpose_bf16_kernel<<<g, 256, 0, stream>>>(rkernel_f, rktf, UQ, GD);
    int n2 = UQ * UQ;
    transpose_bf16_kernel<<<(n2 + 255) / 256, 256, 0, stream>>>(gtran, gtt, UQ, UQ);
  }
  // 2) embedding gather (padded, bf16)
  {
    size_t n = (size_t)BQ * LP * EQ;
    embed_kernel<<<(unsigned)((n + 255) / 256), 256, 0, stream>>>(ids, emb, x0);
  }
  // 3) padded bf16 adjacency
  {
    size_t n = (size_t)BQ * LP * LP;
    pad_graphs_kernel<<<(unsigned)((n + 255) / 256), 256, 0, stream>>>(graphs, gpad);
  }
  // 4) zero the transposed layer-2 sequence (padded tail multiplies into valid rows)
  {
    size_t n = (size_t)BQ * UQ * LP;
    zero_u16_kernel<<<(unsigned)((n + 255) / 256), 256, 0, stream>>>(x2T, n);
  }

  const dim3 projGrid(BQ * LP / 16, (GD / 16 + 3) / 4, 1);   // 1792 x 17
  const dim3 scanGrid(BQ / 16, 1, 1);                        // 8 persistent WGs

  // layer 0
  wmma_gemm_kernel<<<projGrid, 128, 0, stream>>>(x0, EQ, 0, kt0, EQ, 0, bias0,
                                                 xkb, GD, 0, GD / 16, EQ);
  onlstm_scan_kernel<<<scanGrid, 256, 0, stream>>>(xkb, rkt0, x1, nullptr, nullptr);
  // layer 1 (also emits transposed sequence for graph conv)
  wmma_gemm_kernel<<<projGrid, 128, 0, stream>>>(x1, UQ, 0, kt1, UQ, 0, bias1,
                                                 xkb, GD, 0, GD / 16, UQ);
  onlstm_scan_kernel<<<scanGrid, 256, 0, stream>>>(xkb, rkt1, x2, x2T, nullptr);
  // graph conv: g1[b] = graphs[b] @ x2[b]  (batched), then g = g1 @ graph_tran
  {
    dim3 grid1(LP / 16, (UQ / 16 + 3) / 4, BQ);
    wmma_gemm_kernel<<<grid1, 128, 0, stream>>>(gpad, LP, (size_t)LP * LP,
                                                x2T, LP, (size_t)UQ * LP, nullptr,
                                                g1, UQ, (size_t)LP * UQ, UQ / 16, LP);
    dim3 grid2(BQ * LP / 16, (UQ / 16 + 3) / 4, 1);
    wmma_gemm_kernel<<<grid2, 128, 0, stream>>>(g1, UQ, 0, gtt, UQ, 0, nullptr,
                                                gbf, UQ, 0, UQ / 16, UQ);
  }
  // final ON-LSTM on x2 -> out[:, :256]
  wmma_gemm_kernel<<<projGrid, 128, 0, stream>>>(x2, UQ, 0, ktf, UQ, 0, bias_f,
                                                 xkb, GD, 0, GD / 16, UQ);
  onlstm_scan_kernel<<<scanGrid, 256, 0, stream>>>(xkb, rktf, nullptr, nullptr, out);
  // final ON-LSTM on g -> out[:, 256:]
  wmma_gemm_kernel<<<projGrid, 128, 0, stream>>>(gbf, UQ, 0, ktf, UQ, 0, bias_f,
                                                 xkb, GD, 0, GD / 16, UQ);
  onlstm_scan_kernel<<<scanGrid, 256, 0, stream>>>(xkb, rktf, nullptr, nullptr, out + UQ);
}
